// LSTMCell_22093311770693
// MI455X (gfx1250) — matrix operand order repaired and checked by reading, not compiled, not run on hardware
//
#include <hip/hip_runtime.h>
#include <hip/hip_bf16.h>
#include <math.h>

#define B_ROWS 4096
#define K_DIM  2048   // ISIZE + OSIZE
#define N_DIM  4096   // 4 * OSIZE
#define OSZ    1024
#define LN_EPS 1e-5f

typedef __bf16 bf16_t;
typedef __attribute__((ext_vector_type(16))) __bf16 v16bf;
typedef __attribute__((ext_vector_type(8)))  __bf16 v8bf;
typedef __attribute__((ext_vector_type(8)))  float  v8f;
typedef __attribute__((ext_vector_type(4)))  float  v4f;

// ---------------------------------------------------------------------------
// Pack [x | h] (fp32, row-major 4096 x 2048) into split bf16 hi/lo planes.
// ---------------------------------------------------------------------------
__global__ void __launch_bounds__(256)
lstm_pack_A(const float* __restrict__ x, const float* __restrict__ h,
            bf16_t* __restrict__ ahi, bf16_t* __restrict__ alo) {
    const int idx = (blockIdx.x * 256 + threadIdx.x) * 4;   // element index
    const int r = idx / K_DIM;
    const int k = idx % K_DIM;
    const float* src = (k < OSZ) ? (x + (size_t)r * OSZ + k)
                                 : (h + (size_t)r * OSZ + (k - OSZ));
    v4f v = *(const v4f*)src;           // never straddles the x/h boundary
#pragma unroll
    for (int i = 0; i < 4; ++i) {
        bf16_t hi = (bf16_t)v[i];
        bf16_t lo = (bf16_t)(v[i] - (float)hi);
        ahi[idx + i] = hi;
        alo[idx + i] = lo;
    }
}

// ---------------------------------------------------------------------------
// Pack W (fp32, row-major 4096 x 2048) into split bf16 hi/lo planes.
// ---------------------------------------------------------------------------
__global__ void __launch_bounds__(256)
lstm_pack_W(const float* __restrict__ W,
            bf16_t* __restrict__ whi, bf16_t* __restrict__ wlo) {
    const int idx = (blockIdx.x * 256 + threadIdx.x) * 4;
    v4f v = *(const v4f*)(W + idx);
#pragma unroll
    for (int i = 0; i < 4; ++i) {
        bf16_t hi = (bf16_t)v[i];
        bf16_t lo = (bf16_t)(v[i] - (float)hi);
        whi[idx + i] = hi;
        wlo[idx + i] = lo;
    }
}

// ---------------------------------------------------------------------------
// comb = A @ W^T via split-bf16 "x3" WMMA (Ahi*Bhi + Ahi*Blo + Alo*Bhi).
// Each wave32 computes a 32(M) x 64(N) output tile; K-loop steps of 32.
// A fragment layout (16-bit 16x32): lane<16 holds row (m0+lane), K chunks
// [k0..k0+7] and [k0+16..k0+23]; lane>=16 holds [k0+8..+15], [k0+24..+31].
// B fragment: lane<16 holds column n0+lane, K [k0..k0+7], [k0+16..k0+23]; lane>=16 holds
// column n0+lane-16, K [k0+8..+15], [k0+24..+31] (same order as A) -> rows of W.
// ---------------------------------------------------------------------------
__global__ void __launch_bounds__(256)
lstm_gemm_bf16x3(const bf16_t* __restrict__ Ahi, const bf16_t* __restrict__ Alo,
                 const bf16_t* __restrict__ Whi, const bf16_t* __restrict__ Wlo,
                 float* __restrict__ comb) {
    const int wid  = threadIdx.x >> 5;
    const int lane = threadIdx.x & 31;
    const int gw   = blockIdx.x * 8 + wid;         // 0 .. 8191
    const int NGRP = N_DIM / 64;                   // 64 column groups
    const int m0   = (gw / NGRP) * 32;             // 128 row strips of 32
    const int n0   = (gw % NGRP) * 64;
    const int half = lane >> 4;
    const int lr   = lane & 15;

    v8f acc[2][4];
#pragma unroll
    for (int mt = 0; mt < 2; ++mt)
#pragma unroll
        for (int nt = 0; nt < 4; ++nt)
            acc[mt][nt] = (v8f){0.f, 0.f, 0.f, 0.f, 0.f, 0.f, 0.f, 0.f};

    for (int k0 = 0; k0 < K_DIM; k0 += 32) {
        const int ka = k0 + half * 8;              // A per-lane chunk base
        v16bf a_hi[2], a_lo[2];
#pragma unroll
        for (int mt = 0; mt < 2; ++mt) {
            const size_t rowoff = (size_t)(m0 + mt * 16 + lr) * K_DIM + ka;
            union { v8bf c[2]; v16bf v; } uh, ul;
            uh.c[0] = *(const v8bf*)(Ahi + rowoff);
            uh.c[1] = *(const v8bf*)(Ahi + rowoff + 16);
            ul.c[0] = *(const v8bf*)(Alo + rowoff);
            ul.c[1] = *(const v8bf*)(Alo + rowoff + 16);
            a_hi[mt] = uh.v;
            a_lo[mt] = ul.v;
        }

        const int kb = k0 + half * 8;              // B per-lane chunk base
#pragma unroll
        for (int nt = 0; nt < 4; ++nt) {
            const size_t coloff = (size_t)(n0 + nt * 16 + lr) * K_DIM + kb;
            union { v8bf c[2]; v16bf v; } bh, bl;
            bh.c[0] = *(const v8bf*)(Whi + coloff);
            bh.c[1] = *(const v8bf*)(Whi + coloff + 16);
            bl.c[0] = *(const v8bf*)(Wlo + coloff);
            bl.c[1] = *(const v8bf*)(Wlo + coloff + 16);
            v16bf b_hi = bh.v;
            v16bf b_lo = bl.v;
#pragma unroll
            for (int mt = 0; mt < 2; ++mt) {
                acc[mt][nt] = __builtin_amdgcn_wmma_f32_16x16x32_bf16(
                    false, a_hi[mt], false, b_hi, (short)0, acc[mt][nt], false, false);
                acc[mt][nt] = __builtin_amdgcn_wmma_f32_16x16x32_bf16(
                    false, a_hi[mt], false, b_lo, (short)0, acc[mt][nt], false, false);
                acc[mt][nt] = __builtin_amdgcn_wmma_f32_16x16x32_bf16(
                    false, a_lo[mt], false, b_hi, (short)0, acc[mt][nt], false, false);
            }
        }
    }

    // D layout (32-bit 16x16): VGPR r, lane<16 -> M=r,N=lane; lane>=16 -> M=8+r.
#pragma unroll
    for (int mt = 0; mt < 2; ++mt)
#pragma unroll
        for (int nt = 0; nt < 4; ++nt)
#pragma unroll
            for (int r = 0; r < 8; ++r) {
                const int row = m0 + mt * 16 + half * 8 + r;
                const int col = n0 + nt * 16 + lr;
                comb[(size_t)row * N_DIM + col] = acc[mt][nt][r];
            }
}

// ---------------------------------------------------------------------------
// Per-row LayerNorm over all 4*OSZ values + gates + elementwise LSTM update.
// One block (256 threads) per batch row; each thread owns 4 j-columns x 4 gates.
// ---------------------------------------------------------------------------
__device__ __forceinline__ float sigmoidf_(float v) { return 1.0f / (1.0f + expf(-v)); }
__device__ __forceinline__ float gelu_exact_(float v) {
    return 0.5f * v * (1.0f + erff(v * 0.70710678118654752f));
}

__global__ void __launch_bounds__(256)
lstm_ln_act(const float* __restrict__ comb, const float* __restrict__ c,
            const float* __restrict__ ln_w, const float* __restrict__ ln_b,
            float* __restrict__ out, float* __restrict__ cell) {
    __shared__ float s_sum[256];
    __shared__ float s_ss[256];

    const int row = blockIdx.x;
    const int t   = threadIdx.x;
    const int j0  = t * 4;                         // 256 * 4 = 1024 = OSZ
    const float* rowp = comb + (size_t)row * N_DIM;

    v4f v[4];
    float sum = 0.f, ss = 0.f;
#pragma unroll
    for (int g = 0; g < 4; ++g) {
        v[g] = *(const v4f*)(rowp + g * OSZ + j0);
#pragma unroll
        for (int i = 0; i < 4; ++i) { sum += v[g][i]; ss += v[g][i] * v[g][i]; }
    }
    s_sum[t] = sum;
    s_ss[t]  = ss;
    __syncthreads();
    for (int off = 128; off > 0; off >>= 1) {
        if (t < off) { s_sum[t] += s_sum[t + off]; s_ss[t] += s_ss[t + off]; }
        __syncthreads();
    }
    const float mean = s_sum[0] * (1.0f / N_DIM);
    const float var  = s_ss[0] * (1.0f / N_DIM) - mean * mean;
    const float rstd = rsqrtf(var + LN_EPS);

    v4f n[4];
#pragma unroll
    for (int g = 0; g < 4; ++g)
#pragma unroll
        for (int i = 0; i < 4; ++i)
            n[g][i] = (v[g][i] - mean) * rstd * ln_w[g * OSZ + j0 + i]
                      + ln_b[g * OSZ + j0 + i];

    v4f cv = *(const v4f*)(c + (size_t)row * OSZ + j0);
    v4f outv, cellv;
#pragma unroll
    for (int i = 0; i < 4; ++i) {
        const float fg  = sigmoidf_(n[0][i]);
        const float og  = sigmoidf_(n[1][i]);
        const float ig  = sigmoidf_(n[2][i]);
        const float hid = gelu_exact_(n[3][i]);
        const float cl  = fg * cv[i] + ig * hid;
        cellv[i] = cl;
        outv[i]  = og * cl;
    }
    *(v4f*)(out  + (size_t)row * OSZ + j0) = outv;
    *(v4f*)(cell + (size_t)row * OSZ + j0) = cellv;
}

// ---------------------------------------------------------------------------
extern "C" void kernel_launch(void* const* d_in, const int* in_sizes, int n_in,
                              void* d_out, int out_size, void* d_ws, size_t ws_size,
                              hipStream_t stream) {
    const float* x    = (const float*)d_in[0];
    const float* h    = (const float*)d_in[1];
    const float* c    = (const float*)d_in[2];
    const float* W    = (const float*)d_in[3];
    const float* ln_w = (const float*)d_in[4];
    const float* ln_b = (const float*)d_in[5];

    float* out  = (float*)d_out;                       // first  B*OSZ floats
    float* cell = out + (size_t)B_ROWS * OSZ;          // second B*OSZ floats

    const size_t AB = (size_t)B_ROWS * K_DIM;          // 8M elements
    bf16_t* Ahi = (bf16_t*)d_ws;
    bf16_t* Alo = Ahi + AB;
    bf16_t* Whi = Alo + AB;
    bf16_t* Wlo = Whi + AB;
    float*  comb = (float*)(Wlo + AB);                 // 4096 x 4096 fp32

    const int packBlocks = (int)(AB / (256 * 4));      // 8192
    lstm_pack_A<<<packBlocks, 256, 0, stream>>>(x, h, Ahi, Alo);
    lstm_pack_W<<<packBlocks, 256, 0, stream>>>(W, Whi, Wlo);

    // 8192 waves total: 128 row-strips (32 rows) x 64 col-groups (64 cols)
    lstm_gemm_bf16x3<<<1024, 256, 0, stream>>>(Ahi, Alo, Whi, Wlo, comb);

    lstm_ln_act<<<B_ROWS, 256, 0, stream>>>(comb, c, ln_w, ln_b, out, cell);
}
